// Decoder_27092653703740
// MI455X (gfx1250) — matrix operand order repaired and checked by reading, not compiled, not run on hardware
//
#include <hip/hip_runtime.h>
#include <hip/hip_bf16.h>
#include <math.h>

typedef unsigned short u16;
typedef __attribute__((ext_vector_type(16))) __bf16 v16bf;
typedef __attribute__((ext_vector_type(8)))  float  v8f;

#define DEV static __device__ __forceinline__

union Frag  { v16bf v; u16 h[16]; uint4 q[2]; };
union Pack8 { uint4 q; u16 h[8]; };

DEV u16 f2bf(float f) {
  union { float f; unsigned u; } x; x.f = f;
  unsigned r = x.u + 0x7FFFu + ((x.u >> 16) & 1u);
  return (u16)(r >> 16);
}
DEV float elu1(float x)  { return x > 0.f ? x : (expf(x) - 1.f); }
DEV float geluE(float x) { return 0.5f * x * (1.f + erff(x * 0.70710678118654752f)); }

// ---------------------------------------------------------------------------
// Generic bf16 WMMA GEMM: C[M,N] = act(A[M,K] @ Wt[N,K]^T + bias)
// one wave computes a 16x16 tile; K multiple of 32.
// ---------------------------------------------------------------------------
__global__ void k_gemm(const u16* __restrict__ A, const u16* __restrict__ Wt,
                       const float* __restrict__ bias,
                       float* outF, u16* outB,
                       int M, int N, int K, int ldOut, int colOff, int act, int Nreal)
{
  int wid  = (int)((blockIdx.x * blockDim.x + threadIdx.x) >> 5);
  int lane = threadIdx.x & 31;
  int nt = N >> 4;
  int tiles = (M >> 4) * nt;
  if (wid >= tiles) return;
  int mt = wid / nt, ntile = wid % nt;
  int lo = lane & 15, hi = lane >> 4;
  int rowA = mt * 16 + lo;
  int colB = ntile * 16 + lo;
  const u16* aRow = A  + (size_t)rowA * K;
  const u16* bRow = Wt + (size_t)colB * K;
  v8f acc; for (int i = 0; i < 8; ++i) acc[i] = 0.f;
  for (int k0 = 0; k0 < K; k0 += 32) {
    Frag a, b;
    const uint4* ap = (const uint4*)(aRow + k0);
    a.q[0] = ap[hi];      // elems 0-7  : K = k0 + hi*8 .. +7
    a.q[1] = ap[2 + hi];  // elems 8-15 : K = k0 + 16 + hi*8 ..
    const uint4* bp = (const uint4*)(bRow + k0);
    b.q[0] = bp[hi]; b.q[1] = bp[2 + hi];   // same K order as A
    acc = __builtin_amdgcn_wmma_f32_16x16x32_bf16(false, a.v, false, b.v,
                                                  (short)0, acc, false, false);
  }
  int n = colB;
  if (n < Nreal) {
    float bv = bias ? bias[n] : 0.f;
    for (int r = 0; r < 8; ++r) {
      int m = mt * 16 + r + hi * 8;
      float vv = acc[r] + bv;
      if (act == 1) vv = geluE(vv);
      else if (act == 2) vv = elu1(vv);
      if (outF) outF[(size_t)m * ldOut + colOff + n] = vv;
      if (outB) outB[(size_t)m * ldOut + colOff + n] = f2bf(vv);
    }
  }
}

// ---------------------------------------------------------------------------
// Flash attention, one wave per (b, n, head, 16-query tile).
// Computes S^T = K @ Q^T (softmax within lane pairs), O^T = V^T @ P^T.
// ---------------------------------------------------------------------------
__global__ void k_attn(const u16* __restrict__ Q, const u16* __restrict__ Kb,
                       const u16* __restrict__ Vt, u16* __restrict__ Out,
                       const float* __restrict__ tq, const float* __restrict__ valid,
                       int Sk, int mode)
{
  int wid  = (int)((blockIdx.x * blockDim.x + threadIdx.x) >> 5);
  if (wid >= 8 * 10 * 8 * 15) return;
  int lane = threadIdx.x & 31;
  int qt = wid % 15; int t = wid / 15;
  int h = t % 8; t /= 8;
  int n = t % 10; int b = t / 10;
  int lo = lane & 15, hi = lane >> 4;
  int qg = qt * 16 + lo;
  size_t qrow = (size_t)(b * 10 + n) * 240 + qg;

  Frag qf;  // B operand: Q^T, lane=query col, K(=hd) in table order (runs of 8)
  { const uint4* p = (const uint4*)(Q + qrow * 256 + h * 32);
    qf.q[0] = p[hi]; qf.q[1] = p[2 + hi]; }

  v8f o0, o1, zc;
  for (int i = 0; i < 8; ++i) { o0[i] = 0.f; o1[i] = 0.f; zc[i] = 0.f; }
  float m = -3e30f, lsum = 0.f;
  float tqv = tq[b * 240 + qg];
  const float scale = 0.17677669529663687f; // 1/sqrt(32)
  int brq = qg >> 3, tsq = qg & 7;
  int nkb = (Sk + 31) >> 5;

  for (int kb = 0; kb < nkb; ++kb) {
    Frag k0f, k1f;
    for (int tt = 0; tt < 2; ++tt) {
      Frag* kf = tt ? &k1f : &k0f;
      int key = kb * 32 + tt * 16 + lo;
      if (key < Sk) {
        const uint4* p = (const uint4*)(Kb + (size_t)(b * Sk + key) * 256 + h * 32);
        kf->q[0] = p[hi]; kf->q[1] = p[2 + hi];
      } else {
        kf->q[0] = make_uint4(0, 0, 0, 0); kf->q[1] = make_uint4(0, 0, 0, 0);
      }
    }
    v8f s0 = __builtin_amdgcn_wmma_f32_16x16x32_bf16(false, k0f.v, false, qf.v, (short)0, zc, false, false);
    v8f s1 = __builtin_amdgcn_wmma_f32_16x16x32_bf16(false, k1f.v, false, qf.v, (short)0, zc, false, false);

    float p0[8], p1[8];
    float mloc = -3e30f;
    for (int r = 0; r < 8; ++r) {
      int kg0 = kb * 32 + hi * 8 + r;   // S0 tile rows
      int kg1 = kg0 + 16;               // S1 tile rows
      float mv0, mv1;
      if (mode == 0) {
        mv0 = (kg0 < Sk && tqv * valid[b * 256 + kg0] == 1.f) ? 0.f : -1e9f;
        mv1 = (kg1 < Sk && tqv * valid[b * 256 + kg1] == 1.f) ? 0.f : -1e9f;
      } else {
        mv0 = (kg0 < Sk && (kg0 >> 3) == brq && (kg0 & 7) <= tsq && tqv == 1.f) ? 0.f : -1e9f;
        mv1 = (kg1 < Sk && (kg1 >> 3) == brq && (kg1 & 7) <= tsq && tqv == 1.f) ? 0.f : -1e9f;
      }
      p0[r] = s0[r] * scale + mv0;
      p1[r] = s1[r] * scale + mv1;
      mloc = fmaxf(mloc, fmaxf(p0[r], p1[r]));
    }
    mloc = fmaxf(mloc, __shfl_xor(mloc, 16, 32));
    float mnew  = fmaxf(m, mloc);
    float alpha = expf(m - mnew);
    float psum = 0.f;
    Frag pf;  // B operand P^T: lane=query col, K(=key) = hi*16 + e
    for (int r = 0; r < 8; ++r) {
      float e0 = expf(p0[r] - mnew);
      float e1 = expf(p1[r] - mnew);
      psum += e0 + e1;
      float o0v = __shfl_xor(e0, 16, 32);
      float o1v = __shfl_xor(e1, 16, 32);
      if (hi == 0) { pf.h[r] = f2bf(e0);  pf.h[8 + r] = f2bf(o0v); }
      else         { pf.h[r] = f2bf(o1v); pf.h[8 + r] = f2bf(e1); }
    }
    psum += __shfl_xor(psum, 16, 32);
    lsum = lsum * alpha + psum;
    m = mnew;
    for (int i = 0; i < 8; ++i) { o0[i] *= alpha; o1[i] *= alpha; }

    Frag v0f, v1f;  // A operand V^T: lane=hd row, keys = hi*16 + e
    const u16* vb0 = Vt + ((size_t)(b * 8 + h) * 32 + lo) * 256 + kb * 32;
    const u16* vb1 = vb0 + 16 * 256;
    { const uint4* p = (const uint4*)vb0; v0f.q[0] = p[2 * hi]; v0f.q[1] = p[2 * hi + 1]; }
    { const uint4* p = (const uint4*)vb1; v1f.q[0] = p[2 * hi]; v1f.q[1] = p[2 * hi + 1]; }
    o0 = __builtin_amdgcn_wmma_f32_16x16x32_bf16(false, v0f.v, false, pf.v, (short)0, o0, false, false);
    o1 = __builtin_amdgcn_wmma_f32_16x16x32_bf16(false, v1f.v, false, pf.v, (short)0, o1, false, false);
  }
  float inv = 1.f / lsum;
  Pack8 st;
  u16* ob = Out + qrow * 256 + h * 32 + hi * 8;
  for (int r = 0; r < 8; ++r) st.h[r] = f2bf(o0[r] * inv);
  *(uint4*)ob = st.q;
  for (int r = 0; r < 8; ++r) st.h[r] = f2bf(o1[r] * inv);
  *(uint4*)(ob + 16) = st.q;
}

// ---------------------------------------------------------------------------
// Support kernels
// ---------------------------------------------------------------------------
__global__ void k_wconv(const float* __restrict__ W, u16* __restrict__ Wt,
                        int K, int N, int Npad)
{
  int i = blockIdx.x * blockDim.x + threadIdx.x;
  if (i >= Npad * K) return;
  int n = i / K, k = i % K;
  Wt[i] = (n < N) ? f2bf(W[(size_t)k * N + n]) : (u16)0;
}

__global__ void k_tobf(const float* __restrict__ X, u16* __restrict__ Y, int nElem)
{
  int i = blockIdx.x * blockDim.x + threadIdx.x;
  if (i < nElem) Y[i] = f2bf(X[i]);
}

__global__ void k_ete(const float* __restrict__ X, const float* __restrict__ W1,
                      const float* __restrict__ B1, const float* __restrict__ W2,
                      const float* __restrict__ B2, float* __restrict__ Ori)
{
  __shared__ float sx[6]; __shared__ float sh[64];
  int row = blockIdx.x; int c = threadIdx.x;
  if (c < 6) sx[c] = X[(size_t)row * 6 + c];
  __syncthreads();
  if (c < 64) {
    float a = B1[c];
    for (int i = 0; i < 6; ++i) a += sx[i] * W1[i * 64 + c];
    sh[c] = a > 0.f ? a : 0.f;
  }
  __syncthreads();
  float a = B2[c];
  for (int j = 0; j < 64; ++j) a += sh[j] * W2[j * 256 + c];
  Ori[(size_t)row * 256 + c] = a;
}

__global__ void k_pool(const float* __restrict__ Ori, u16* __restrict__ Pb)
{
  int i = blockIdx.x * blockDim.x + threadIdx.x;
  if (i >= 8 * 240 * 256) return;
  int c = i & 255; int row = i >> 8;
  int b = row / 240, qg = row % 240;
  int br = qg >> 3, ts = qg & 7;
  float mx = -3e30f;
  size_t base = ((size_t)(b * 30 + br) * 80 + ts * 10) * 256 + c;
  for (int j = 0; j < 10; ++j) mx = fmaxf(mx, Ori[base + (size_t)j * 256]);
  Pb[i] = f2bf(mx);
}

__global__ void k_tq(const float* __restrict__ X, float* __restrict__ Tq)
{
  int i = blockIdx.x * blockDim.x + threadIdx.x;
  if (i >= 1920) return;
  int b = i / 240, qg = i % 240; int br = qg >> 3, ts = qg & 7;
  const float* p = X + ((size_t)(b * 30 + br) * 80 + ts * 10) * 6;
  float s = 0; for (int c = 0; c < 6; ++c) s += p[c];
  Tq[i] = (s != 0.f) ? 1.f : 0.f;
}

__global__ void k_valid(const int* __restrict__ Mk, float* __restrict__ V)
{
  int i = blockIdx.x * blockDim.x + threadIdx.x;
  if (i < 2048) V[i] = Mk[i] ? 0.f : 1.f;
}

__global__ void k_query(const float* __restrict__ Enc, const float* __restrict__ Temb,
                        const float* __restrict__ Ori, const int* __restrict__ Ts,
                        u16* __restrict__ Qy)
{
  int i = blockIdx.x * blockDim.x + threadIdx.x;
  if (i >= 8 * 10 * 240 * 256) return;
  int c = i & 255; int row = i >> 8;
  int qg = row % 240; int t2 = row / 240;
  int n = t2 % 10; int b = t2 / 10;
  int br = qg >> 3, ts = qg & 7;
  int T = Ts[0];
  float v = Enc[((size_t)b * 256 + 1 + n) * 256 + c]
          + Temb[ts * 256 + c]
          + Ori[((size_t)(b * 30 + br) * 80 + (T - 1)) * 256 + c];
  Qy[i] = f2bf(v);
}

__global__ void k_vt(const u16* __restrict__ V, u16* __restrict__ Vt, int Sk)
{
  int i = blockIdx.x * blockDim.x + threadIdx.x;
  if (i >= 8 * 8 * 32 * 256) return;
  int key = i & 255; int hd = (i >> 8) & 31; int h = (i >> 13) & 7; int b = i >> 16;
  u16 v = 0;
  if (key < Sk) v = V[(size_t)(b * Sk + key) * 256 + h * 32 + hd];
  Vt[i] = v;
}

__global__ void k_ln(const float* __restrict__ X, const float* __restrict__ Res,
                     const float* __restrict__ G, const float* __restrict__ Bt,
                     float* outF, u16* outB, int ldB, int coffB)
{
  int row = blockIdx.x; int c = threadIdx.x;
  float x = X[(size_t)row * 256 + c];
  if (Res) x += Res[(size_t)row * 256 + c];
  __shared__ float red[8];
  float v = x;
  for (int o = 16; o; o >>= 1) v += __shfl_xor(v, o, 32);
  if ((c & 31) == 0) red[c >> 5] = v;
  __syncthreads();
  float mu = 0.f; for (int i = 0; i < 8; ++i) mu += red[i];
  mu *= (1.f / 256.f);
  __syncthreads();
  float d = x - mu; v = d * d;
  for (int o = 16; o; o >>= 1) v += __shfl_xor(v, o, 32);
  if ((c & 31) == 0) red[c >> 5] = v;
  __syncthreads();
  float var = 0.f; for (int i = 0; i < 8; ++i) var += red[i];
  var *= (1.f / 256.f);
  float y = d * rsqrtf(var + 1e-5f) * G[c] + Bt[c];
  if (outF) outF[(size_t)row * 256 + c] = y;
  if (outB) outB[(size_t)row * ldB + coffB + c] = f2bf(y);
}

__global__ void k_traj(const float* __restrict__ Raw, const float* __restrict__ AS,
                       float* __restrict__ Out)
{
  int i = blockIdx.x * blockDim.x + threadIdx.x;
  if (i >= 576000) return;
  int c = i % 3; int t = (i / 3) % 80; int n = (i / 240) % 10;
  int br = (i / 2400) % 30; int b = i / 72000;
  int row = (b * 10 + n) * 240 + br * 8 + (t / 10);
  int col = (t % 10) * 3 + c;
  float cs = AS[((size_t)(b * 10 + n) * 11 + 10) * 11 + c];
  Out[i] = Raw[(size_t)row * 32 + col] + cs;
}

__global__ void k_scfeat(const float* __restrict__ E, const float* __restrict__ AT,
                         const float* __restrict__ AS,
                         const float* __restrict__ W1, const float* __restrict__ B1,
                         const float* __restrict__ W2, const float* __restrict__ B2,
                         float* __restrict__ Mf)
{
  int idx = blockIdx.x;                 // (b,br,t)
  int t = idx % 80; int br = (idx / 80) % 30; int b = idx / 2400;
  int j = threadIdx.x;                  // 0..63
  __shared__ float h1[64];
  const float* e = E + ((size_t)(b * 30 + br) * 80 + t) * 6;
  float ex = e[0], ey = e[1], eyaw = e[2], esp = e[3];
  float evx = esp * cosf(eyaw), evy = esp * sinf(eyaw);
  float mx = -3e30f;
  for (int n = 0; n < 10; ++n) {
    const float* abase = AT + ((size_t)((b * 30 + br) * 10 + n) * 80) * 3;
    const float* a = abase + t * 3;
    float ax = a[0], ay = a[1], ayaw = a[2];
    int tv = t > 0 ? t : 1;
    float avx = (abase[tv * 3 + 0] - abase[(tv - 1) * 3 + 0]) * 10.f;
    float avy = (abase[tv * 3 + 1] - abase[(tv - 1) * 3 + 1]) * 10.f;
    float ry0 = ayaw - eyaw;
    float ry = atan2f(sinf(ry0), cosf(ry0));
    float cr = cosf(ry), sr = sinf(ry);
    const float* cs = AS + ((size_t)(b * 10 + n) * 11 + 10) * 11;
    float msum = 0.f; for (int k = 0; k < 11; ++k) msum += cs[k];
    float am = (msum != 0.f) ? 1.f : 0.f;
    float at10[10];
    at10[0] = (ax - ex) * cr; at10[1] = (ay - ey) * sr; at10[2] = ry;
    at10[3] = (avx - evx) * cr; at10[4] = (avy - evy) * sr;
    at10[5] = cs[6]; at10[6] = cs[7]; at10[7] = cs[8]; at10[8] = cs[9]; at10[9] = cs[10];
    float a1 = B1[j];
    for (int i2 = 0; i2 < 10; ++i2) a1 += at10[i2] * am * W1[i2 * 64 + j];
    h1[j] = elu1(a1);
    __syncthreads();
    float f = B2[j];
    for (int k = 0; k < 64; ++k) f += h1[k] * W2[k * 64 + j];
    mx = fmaxf(mx, f);
    __syncthreads();
  }
  Mf[(size_t)idx * 64 + j] = mx;
}

__global__ void k_featmean(const float* __restrict__ Mf, const int* __restrict__ Ts,
                           float* __restrict__ Fm)
{
  int i = blockIdx.x * blockDim.x + threadIdx.x;
  if (i >= 240 * 64) return;
  int j = i & 63; int rb = i >> 6;
  int mt = Ts[0] * 10;
  float s = 0.f;
  for (int t = 0; t < mt; ++t) s += Mf[((size_t)rb * 80 + t) * 64 + j];
  Fm[i] = s / (float)mt;
}

__global__ void k_inter(const float* __restrict__ Fm,
                        const float* __restrict__ W1, const float* __restrict__ B1,
                        const float* __restrict__ W2, const float* __restrict__ B2,
                        float* __restrict__ Itx)
{
  int i = blockIdx.x * blockDim.x + threadIdx.x;
  if (i >= 240) return;
  float h[32];
  for (int j = 0; j < 32; ++j) {
    float a = B1[j];
    for (int k = 0; k < 64; ++k) a += Fm[i * 64 + k] * W1[k * 32 + j];
    h[j] = elu1(a);
  }
  for (int o = 0; o < 4; ++o) {
    float a = B2[o];
    for (int j = 0; j < 32; ++j) a += h[j] * W2[j * 4 + o];
    Itx[i * 4 + o] = 1.f / (1.f + expf(-a));
  }
}

__global__ void k_hard(const float* __restrict__ E, const int* __restrict__ Ts,
                       float* __restrict__ Hc)
{
  int i = blockIdx.x * blockDim.x + threadIdx.x;
  if (i >= 240) return;
  int mt = Ts[0] * 10;
  const float* p = E + (size_t)i * 80 * 6;
  float ss = 0, sa = 0, sj = 0, sl = 0;
  float accPrev = p[4];
  float acc1 = p[1 * 6 + 4];
  for (int t = 0; t < mt; ++t) {
    float sp = p[t * 6 + 3], ac = p[t * 6 + 4], cv = p[t * 6 + 5];
    ss += sp; sa += fabsf(ac);
    float jk = (t == 0) ? (acc1 - p[4]) * 10.f : (ac - accPrev) * 10.f;
    sj += fabsf(jk);
    accPrev = ac;
    sl += fabsf(sp * sp * cv);
  }
  float fmt = (float)mt;
  Hc[i * 4 + 0] = fminf(fmaxf(-(ss / fmt), 0.f), 15.f) / 15.f;
  Hc[i * 4 + 1] = fminf(fmaxf(sa / fmt, 0.f), 4.f) / 4.f;
  Hc[i * 4 + 2] = fminf(fmaxf(sj / fmt, 0.f), 6.f) / 6.f;
  Hc[i * 4 + 3] = fminf(fmaxf(sl / fmt, 0.f), 5.f) / 5.f;
}

__global__ void k_wts(const float* __restrict__ W1, const float* __restrict__ B1,
                      const float* __restrict__ W2, const float* __restrict__ B2,
                      float* __restrict__ OutW)
{
  int b = blockIdx.x; int j = threadIdx.x;
  __shared__ float h[64];
  float a = B1[j];
  for (int k = 0; k < 256; ++k) a += W1[k * 64 + j];
  h[j] = elu1(a);
  __syncthreads();
  if (j < 8) {
    float o = B2[j];
    for (int k = 0; k < 64; ++k) o += h[k] * W2[k * 8 + j];
    OutW[b * 8 + j] = (o > 20.f) ? o : log1pf(expf(o));
  }
}

__global__ void k_scores(const float* __restrict__ Hc, const float* __restrict__ Itx,
                         const float* __restrict__ Wt8, const float* __restrict__ E,
                         float* __restrict__ OutS)
{
  int i = blockIdx.x * blockDim.x + threadIdx.x;
  if (i >= 240) return;
  int b = i / 30;
  float s = 0.f;
  for (int k = 0; k < 4; ++k) s += Hc[i * 4 + k] * Wt8[b * 8 + k];
  for (int k = 0; k < 4; ++k) s += Itx[i * 4 + k] * Wt8[b * 8 + 4 + k];
  s = -s;
  const float* p = E + (size_t)i * 480;
  float t = 0.f; for (int k = 0; k < 480; ++k) t += p[k];
  OutS[i] = (t != 0.f) ? s : -1e9f;
}

__global__ void k_reg(const float* __restrict__ Enc, const float* __restrict__ W1,
                      const float* __restrict__ B1, const float* __restrict__ W2,
                      const float* __restrict__ B2, float* __restrict__ OutR)
{
  int b = blockIdx.x; int c = threadIdx.x;
  __shared__ float h[256];
  const float* e = Enc + (size_t)b * 256 * 256;  // encoding[:,0]
  float a = B1[c];
  for (int k = 0; k < 256; ++k) a += e[k] * W1[k * 256 + c];
  h[c] = elu1(a);
  __syncthreads();
  for (int u = c; u < 720; u += 256) {
    float o = B2[u];
    for (int k = 0; k < 256; ++k) o += h[k] * W2[k * 720 + u];
    OutR[b * 720 + u] = o;
  }
}

// ---------------------------------------------------------------------------
extern "C" void kernel_launch(void* const* d_in, const int* in_sizes, int n_in,
                              void* d_out, int out_size, void* d_ws, size_t ws_size,
                              hipStream_t stream)
{
  const float* ego     = (const float*)d_in[0];
  const float* astates = (const float*)d_in[1];
  const float* enc     = (const float*)d_in[2];
  const int*   emask   = (const int*)d_in[3];
  const int*   tsp     = (const int*)d_in[4];
  auto F = [&](int i) { return (const float*)d_in[i]; };

  float* out = (float*)d_out;
  float* OUT_TRAJ = out;
  float* OUT_SC   = out + 576000;
  float* OUT_REG  = out + 576240;
  float* OUT_W    = out + 578160;

  // ---- workspace bump allocator (persist region) ----
  char* base = (char*)d_ws;
  size_t off = 0;
  auto alloc = [&](size_t bytes) -> char* {
    char* p = base + off; off = (off + bytes + 255) & ~(size_t)255; return p;
  };
  u16* wT[2][6]; // per-attention: wq,wk,wv,wo,ffn1,ffn2 (bf16, (Npad,K))
  for (int a = 0; a < 2; ++a) {
    for (int w = 0; w < 4; ++w) wT[a][w] = (u16*)alloc(256 * 256 * 2);
    wT[a][4] = (u16*)alloc(1024 * 256 * 2);
    wT[a][5] = (u16*)alloc(256 * 1024 * 2);
  }
  u16* ad1T  = (u16*)alloc(256 * 512 * 2);
  u16* ad2T  = (u16*)alloc(32 * 256 * 2);
  u16* query = (u16*)alloc((size_t)19200 * 256 * 2);
  u16* encb  = (u16*)alloc((size_t)2048 * 256 * 2);
  u16* poolb = (u16*)alloc((size_t)1920 * 256 * 2);
  float* tqv  = (float*)alloc(1920 * 4);
  float* vldv = (float*)alloc(2048 * 4);
  u16* decat = (u16*)alloc((size_t)19200 * 512 * 2);
  float* featm = (float*)alloc(240 * 64 * 4);
  float* inter = (float*)alloc(240 * 4 * 4);
  float* hardf = (float*)alloc(240 * 4 * 4);

  // ---- scratch region (phase-aliased) ----
  char* scr = alloc(91619328);
  u16*   S_Q    = (u16*)(scr + 0);
  u16*   S_K    = (u16*)(scr + 9830400);
  u16*   S_V    = (u16*)(scr + 10878976);
  u16*   S_Vt   = (u16*)(scr + 11927552);
  u16*   S_attn = (u16*)(scr + 12976128);
  float* S_aw   = (float*)(scr + 22806528);
  u16*   S_ffh  = (u16*)(scr + 0);          // aliases Q..aw (dead by FFN time)
  float* S_outf = (float*)(scr + 42467328);
  u16*   S_outb = (u16*)(scr + 62128128);
  float* S_ffn  = (float*)(scr + 71958528);
  float* S_ori  = (float*)(scr + 0);        // phase A
  u16*   S_ad1  = (u16*)(scr + 0);          // phase C
  float* S_traj = (float*)(scr + 9830400);  // phase C
  float* S_mf   = (float*)(scr + 0);        // phase D

  auto gemm = [&](const u16* A, const u16* Wt, const float* bias, float* oF, u16* oB,
                  int M, int N, int K, int ld, int coff, int act, int Nreal) {
    int tiles = (M / 16) * (N / 16);
    k_gemm<<<dim3((tiles + 3) / 4), dim3(128), 0, stream>>>(A, Wt, bias, oF, oB,
                                                            M, N, K, ld, coff, act, Nreal);
  };
  auto wconv = [&](const float* W, u16* Wt, int K, int N, int Npad) {
    int n = Npad * K;
    k_wconv<<<dim3((n + 255) / 256), dim3(256), 0, stream>>>(W, Wt, K, N, Npad);
  };

  // ---- Phase 0: weights -> bf16 transposed ----
  for (int a = 0; a < 2; ++a) {
    int pb = a ? 26 : 10;
    for (int w = 0; w < 4; ++w) wconv(F(pb + w), wT[a][w], 256, 256, 256);
    wconv(F(pb + 12), wT[a][4], 256, 1024, 1024);
    wconv(F(pb + 14), wT[a][5], 1024, 256, 256);
  }
  wconv(F(42), ad1T, 512, 256, 256);
  wconv(F(44), ad2T, 256, 30, 32);

  // ---- Phase A: ego encoder, pooling, masks, query ----
  k_ete<<<dim3(19200), dim3(256), 0, stream>>>(ego, F(5), F(6), F(7), F(8), S_ori);
  k_pool<<<dim3(1920), dim3(256), 0, stream>>>(S_ori, poolb);
  k_tq<<<dim3(8), dim3(256), 0, stream>>>(ego, tqv);
  k_valid<<<dim3(8), dim3(256), 0, stream>>>(emask, vldv);
  k_tobf<<<dim3(2048), dim3(256), 0, stream>>>(enc, encb, 8 * 256 * 256);
  k_query<<<dim3(19200), dim3(256), 0, stream>>>(enc, F(9), S_ori, tsp, query);

  // ---- Phase B: two cross-attentions ----
  for (int a = 0; a < 2; ++a) {
    int pb = a ? 26 : 10;
    const u16* kvA = a ? poolb : encb;
    int Sk = a ? 240 : 256;
    int Mkv = 8 * Sk;
    gemm(query, wT[a][0], F(pb + 4), nullptr, S_Q, 19200, 256, 256, 256, 0, 0, 256);
    gemm(kvA,   wT[a][1], F(pb + 5), nullptr, S_K, Mkv, 256, 256, 256, 0, 0, 256);
    gemm(kvA,   wT[a][2], F(pb + 6), nullptr, S_V, Mkv, 256, 256, 256, 0, 0, 256);
    k_vt<<<dim3(2048), dim3(256), 0, stream>>>(S_V, S_Vt, Sk);
    k_attn<<<dim3(2400), dim3(128), 0, stream>>>(S_Q, S_K, S_Vt, S_attn, tqv, vldv, Sk, a);
    gemm(S_attn, wT[a][3], F(pb + 7), S_aw, nullptr, 19200, 256, 256, 256, 0, 0, 256);
    k_ln<<<dim3(19200), dim3(256), 0, stream>>>(S_aw, nullptr, F(pb + 8), F(pb + 9),
                                                S_outf, S_outb, 256, 0);
    gemm(S_outb, wT[a][4], F(pb + 13), nullptr, S_ffh, 19200, 1024, 256, 1024, 0, 1, 1024);
    gemm(S_ffh,  wT[a][5], F(pb + 15), S_ffn, nullptr, 19200, 256, 1024, 256, 0, 0, 256);
    k_ln<<<dim3(19200), dim3(256), 0, stream>>>(S_ffn, S_outf, F(pb + 10), F(pb + 11),
                                                nullptr, decat, 512, a * 256);
  }

  // ---- Phase C: agent decoder + trajectory assembly ----
  gemm(decat, ad1T, F(43), nullptr, S_ad1, 19200, 256, 512, 256, 0, 2, 256);
  gemm(S_ad1, ad2T, F(45), S_traj, nullptr, 19200, 32, 256, 32, 0, 0, 30);
  k_traj<<<dim3(2250), dim3(256), 0, stream>>>(S_traj, astates, OUT_TRAJ);

  // ---- Phase D: scorer + regressor ----
  k_scfeat<<<dim3(19200), dim3(64), 0, stream>>>(ego, OUT_TRAJ, astates,
                                                 F(50), F(51), F(52), F(53), S_mf);
  k_featmean<<<dim3(60), dim3(256), 0, stream>>>(S_mf, tsp, featm);
  k_inter<<<dim3(1), dim3(256), 0, stream>>>(featm, F(54), F(55), F(56), F(57), inter);
  k_hard<<<dim3(1), dim3(256), 0, stream>>>(ego, tsp, hardf);
  k_wts<<<dim3(8), dim3(64), 0, stream>>>(F(58), F(59), F(60), F(61), OUT_W);
  k_scores<<<dim3(1), dim3(256), 0, stream>>>(hardf, inter, OUT_W, ego, OUT_SC);
  k_reg<<<dim3(8), dim3(256), 0, stream>>>(enc, F(46), F(47), F(48), F(49), OUT_REG);
}
